// reGAU_40613210751318
// MI455X (gfx1250) — matrix operand order repaired and checked by reading, compiled, not run on hardware
//
#include <hip/hip_runtime.h>
#include <hip/hip_bf16.h>
#include <stdint.h>

#define BB 4
#define TT 12
#define NN 2048
#define FF 64
#define HH 4
#define DD 16
#define CC 64

typedef __attribute__((ext_vector_type(16))) __bf16 v16bf;
typedef __attribute__((ext_vector_type(8)))  float  v8f;

union BfFrag {
    v16bf v;
    unsigned short h[16];
    uint4 q[2];
};

__device__ __forceinline__ unsigned short f2bf(float f) {
    union { float f; unsigned u; } c; c.f = f;
    unsigned u = c.u;
    return (unsigned short)((u + 0x7FFFu + ((u >> 16) & 1u)) >> 16);
}
__device__ __forceinline__ float bf2f(unsigned short h) {
    union { unsigned u; float f; } c; c.u = ((unsigned)h) << 16;
    return c.f;
}
__device__ __forceinline__ float lrelu_bias(float f1v, float f2v, float bv) {
    float s = f1v + f2v;
    s = (s > 0.f) ? s : 0.2f * s;
    return s + bv;
}

// ---------------------------------------------------------------------------
// Kernel 1: X[bt] @ {W_z, W_h, Wg_z, Wg_h} via bf16 WMMA.
//   feats stored TRANSPOSED bf16: fzT/fhT[bt][h][d][n]  (B-fragment friendly)
//   xwz/xwh fp32 [bt][n][c]
// ---------------------------------------------------------------------------
__global__ __launch_bounds__(256) void k_proj(
    const float* __restrict__ X,
    const float* __restrict__ Wz, const float* __restrict__ Wh,
    const float* __restrict__ Wgz, const float* __restrict__ Wgh,
    unsigned short* __restrict__ fzT, unsigned short* __restrict__ fhT,
    float* __restrict__ xwz, float* __restrict__ xwh)
{
    __shared__ unsigned short xl[128 * FF];     // 16 KB, [n][f] bf16
    __shared__ unsigned short wl[4 * CC * FF];  // 32 KB, [mat][c][f] bf16

    const int tid   = threadIdx.x;
    const int bt    = blockIdx.y;
    const int nbase = blockIdx.x * 128;

    // stage 128x64 X tile -> bf16 LDS
    const float* xt = X + ((size_t)bt * NN + nbase) * FF;
    for (int i = 0; i < 8; ++i) {
        int e = i * 1024 + tid * 4;
        float4 v = *(const float4*)(xt + e);
        xl[e]     = f2bf(v.x); xl[e + 1] = f2bf(v.y);
        xl[e + 2] = f2bf(v.z); xl[e + 3] = f2bf(v.w);
    }
    // stage all 4 weight matrices transposed [c][f] -> bf16 LDS
    for (int e = tid; e < 4 * CC * FF; e += 256) {
        int mat = e >> 12, rem = e & 4095, c = rem >> 6, f = rem & 63;
        float w;
        if (mat == 0)      w = Wz[f * CC + c];
        else if (mat == 1) w = Wh[f * CC + c];
        else {
            const float* Wg = (mat == 2) ? Wgz : Wgh;
            w = Wg[(((c >> 4) * FF) + f) * DD + (c & 15)];
        }
        wl[e] = f2bf(w);
    }
    __syncthreads();

    const int wave = tid >> 5, lane = tid & 31;
    const int M    = lane & 15;
    const int hi8  = (lane >= 16) ? 8 : 0;
    const int koff = (lane >= 16) ? 8 : 0;    // A-fragment K sub-offset
    const int boff = (lane >= 16) ? 8 : 0;    // B-fragment K sub-offset
    const int r0   = wave * 16;

    // A fragments (rows of X tile), K-split 0..31 / 32..63
    BfFrag a0, a1;
    const unsigned short* xrow = &xl[(r0 + M) * FF];
    a0.q[0] = *(const uint4*)(xrow + koff);
    a0.q[1] = *(const uint4*)(xrow + koff + 16);
    a1.q[0] = *(const uint4*)(xrow + 32 + koff);
    a1.q[1] = *(const uint4*)(xrow + 32 + koff + 16);

    for (int mat = 0; mat < 4; ++mat) {
        for (int ct = 0; ct < 4; ++ct) {
            const unsigned short* wrow = &wl[(mat * CC + ct * 16 + M) * FF];
            BfFrag b0, b1;
            b0.q[0] = *(const uint4*)(wrow + boff);
            b0.q[1] = *(const uint4*)(wrow + boff + 16);
            b1.q[0] = *(const uint4*)(wrow + 32 + boff);
            b1.q[1] = *(const uint4*)(wrow + 32 + boff + 16);

            v8f acc;
            for (int j = 0; j < 8; ++j) acc[j] = 0.f;
            acc = __builtin_amdgcn_wmma_f32_16x16x32_bf16(
                      false, a0.v, false, b0.v, (short)0, acc, false, false);
            acc = __builtin_amdgcn_wmma_f32_16x16x32_bf16(
                      false, a1.v, false, b1.v, (short)0, acc, false, false);

            if (mat < 2) {
                float* dst = (mat == 0 ? xwz : xwh)
                           + ((size_t)bt * NN + nbase + r0) * CC + ct * 16 + M;
                for (int j = 0; j < 8; ++j) dst[(size_t)(j + hi8) * CC] = acc[j];
            } else {
                int cc = ct * 16 + M;
                unsigned short* dT = (mat == 2 ? fzT : fhT)
                    + ((size_t)bt * HH + (cc >> 4)) * DD * NN
                    + (size_t)(cc & 15) * NN + nbase + r0;
                for (int j = 0; j < 8; ++j) dT[j + hi8] = f2bf(acc[j]);
            }
        }
    }
}

// ---------------------------------------------------------------------------
// Kernel 2: f1/f2 = feats . a1/a2 per head (tiny).
// ---------------------------------------------------------------------------
__global__ __launch_bounds__(256) void k_f1f2(
    const unsigned short* __restrict__ fzT, const unsigned short* __restrict__ fhT,
    const float* __restrict__ a1z, const float* __restrict__ a2z,
    const float* __restrict__ a1h, const float* __restrict__ a2h,
    float* __restrict__ f1z, float* __restrict__ f2z,
    float* __restrict__ f1h, float* __restrict__ f2h)
{
    int idx = blockIdx.x * 256 + threadIdx.x;    // over B*T*H*N
    int n   = idx & (NN - 1);
    int bth = idx >> 11;
    int h   = bth & (HH - 1);
    const unsigned short* pz = fzT + (size_t)bth * DD * NN + n;
    const unsigned short* ph = fhT + (size_t)bth * DD * NN + n;
    float s1z = 0, s2z = 0, s1h = 0, s2h = 0;
#pragma unroll
    for (int d = 0; d < DD; ++d) {
        float vz = bf2f(pz[(size_t)d * NN]);
        float vh = bf2f(ph[(size_t)d * NN]);
        s1z += vz * a1z[h * DD + d];
        s2z += vz * a2z[h * DD + d];
        s1h += vh * a1h[h * DD + d];
        s2h += vh * a2h[h * DD + d];
    }
    f1z[idx] = s1z; f2z[idx] = s2z; f1h[idx] = s1h; f2h[idx] = s2h;
}

// ---------------------------------------------------------------------------
// Kernel 3: fused masked-softmax attention + P@V via bf16 WMMA (flash-style).
// One wave = 16 rows of one (b,t,h,gat) instance.
// ---------------------------------------------------------------------------
__global__ __launch_bounds__(256) void k_attn(
    const float* __restrict__ bias_mat,
    const unsigned short* __restrict__ fzT, const unsigned short* __restrict__ fhT,
    const float* __restrict__ f1z, const float* __restrict__ f2z,
    const float* __restrict__ f1h, const float* __restrict__ f2h,
    const float* __restrict__ bgz, const float* __restrict__ bgh,
    float* __restrict__ gatz, float* __restrict__ gath)
{
    const int bt   = blockIdx.y;
    const int b    = bt / TT;
    const int h    = blockIdx.z & 3;
    const int g    = blockIdx.z >> 2;
    const int wave = threadIdx.x >> 5, lane = threadIdx.x & 31;
    const int M    = lane & 15;
    const bool hi  = lane >= 16;
    const int koff = hi ? 8 : 0;
    const int n0   = blockIdx.x * 128 + wave * 16;
    const int row  = n0 + M;

    const unsigned short* fT = g ? fhT : fzT;
    const float* f1  = g ? f1h : f1z;
    const float* f2  = g ? f2h : f2z;
    const float* bg  = g ? bgh : bgz;
    float* gout      = g ? gath : gatz;

    const size_t bth = (size_t)bt * HH + h;
    const float f1v  = f1[bth * NN + row];
    const float* f2p = f2 + bth * NN;
    const float* brow = bias_mat + ((size_t)b * NN + row) * NN;
    const unsigned short* fTcol = fT + bth * DD * NN + (size_t)M * NN;

    // ---- pass 1: row max of masked logits ----
    float rmax = -3.0e38f;
    for (int mb = 0; mb < NN; mb += 32) {
#pragma unroll
        for (int q = 0; q < 4; ++q) {
            int off = mb + koff + (q & 1) * 4 + (q >> 1) * 16;
            float4 bq = *(const float4*)(brow + off);
            float4 fq = *(const float4*)(f2p + off);
            rmax = fmaxf(rmax, lrelu_bias(f1v, fq.x, bq.x));
            rmax = fmaxf(rmax, lrelu_bias(f1v, fq.y, bq.y));
            rmax = fmaxf(rmax, lrelu_bias(f1v, fq.z, bq.z));
            rmax = fmaxf(rmax, lrelu_bias(f1v, fq.w, bq.w));
        }
    }
    rmax = fmaxf(rmax, __shfl_xor(rmax, 16, 32));

    // ---- pass 2: p = exp(logit - max), P@V via WMMA, row sums ----
    float rsum = 0.f;
    v8f acc;
    for (int j = 0; j < 8; ++j) acc[j] = 0.f;
    const int boff = hi ? 8 : 0;

    for (int mb = 0; mb < NN; mb += 32) {
        BfFrag pa, bb;
#pragma unroll
        for (int q = 0; q < 4; ++q) {
            int off = koff + (q & 1) * 4 + (q >> 1) * 16;
            float4 bq = *(const float4*)(brow + mb + off);
            float4 fq = *(const float4*)(f2p + mb + off);
            int hidx = (q >> 1) * 8 + (q & 1) * 4;
            float p;
            p = __expf(lrelu_bias(f1v, fq.x, bq.x) - rmax); rsum += p; pa.h[hidx + 0] = f2bf(p);
            p = __expf(lrelu_bias(f1v, fq.y, bq.y) - rmax); rsum += p; pa.h[hidx + 1] = f2bf(p);
            p = __expf(lrelu_bias(f1v, fq.z, bq.z) - rmax); rsum += p; pa.h[hidx + 2] = f2bf(p);
            p = __expf(lrelu_bias(f1v, fq.w, bq.w) - rmax); rsum += p; pa.h[hidx + 3] = f2bf(p);
        }
        bb.q[0] = *(const uint4*)(fTcol + mb + boff);
        bb.q[1] = *(const uint4*)(fTcol + mb + boff + 16);
        acc = __builtin_amdgcn_wmma_f32_16x16x32_bf16(
                  false, pa.v, false, bb.v, (short)0, acc, false, false);
    }
    rsum += __shfl_xor(rsum, 16, 32);

    // ---- epilogue: normalize, +bg, ELU, store [bt][n][h*16+d] ----
    const float bgv = bg[h * DD + M];
    float* obase = gout + (size_t)bt * NN * CC + (size_t)h * DD + M;
#pragma unroll
    for (int j = 0; j < 8; ++j) {
        int r = j + (hi ? 8 : 0);
        float s = __shfl(rsum, r, 32);
        float v = acc[j] / s + bgv;
        v = (v > 0.f) ? v : (__expf(v) - 1.f);
        obase[(size_t)(n0 + r) * CC] = v;
    }
}

// ---------------------------------------------------------------------------
// Kernel 4: sequential GRU-style gating over T + BatchNorm epilogue.
// ---------------------------------------------------------------------------
__global__ __launch_bounds__(256) void k_recur(
    const float* __restrict__ gatz, const float* __restrict__ gath,
    const float* __restrict__ xwz,  const float* __restrict__ xwh,
    const float* __restrict__ Zb,   const float* __restrict__ Hb,
    const float* __restrict__ gamma, const float* __restrict__ beta,
    const float* __restrict__ mean,  const float* __restrict__ var,
    float* __restrict__ out)
{
    int idx = blockIdx.x * 256 + threadIdx.x;   // over B*N*C
    int c   = idx & (CC - 1);
    int bn  = idx >> 6;
    int b   = bn >> 11;
    size_t base = (size_t)b * TT * NN * CC + (size_t)(bn & (NN - 1)) * CC + c;
    const float zb = Zb[c], hb = Hb[c];
    float hcur = 0.f;
    for (int t = 0; t < TT; ++t) {
        size_t o = base + (size_t)t * NN * CC;
        float z  = 1.f / (1.f + __expf(-(gatz[o] + xwz[o] + zb + hcur)));
        float tt = tanhf(gath[o] + hcur + xwh[o] + hb);
        hcur = z * hcur + (1.f - z) * tt;
    }
    out[idx] = (hcur - mean[c]) * rsqrtf(var[c] + 1e-3f) * gamma[c] + beta[c];
}

// ---------------------------------------------------------------------------
extern "C" void kernel_launch(void* const* d_in, const int* in_sizes, int n_in,
                              void* d_out, int out_size, void* d_ws, size_t ws_size,
                              hipStream_t stream) {
    const float* X    = (const float*)d_in[0];
    const float* bias = (const float*)d_in[1];
    const float* Wz   = (const float*)d_in[2];
    const float* Zb   = (const float*)d_in[3];
    const float* Wh   = (const float*)d_in[4];
    const float* Hb   = (const float*)d_in[5];
    const float* Wgz  = (const float*)d_in[6];
    const float* a1z  = (const float*)d_in[7];
    const float* a2z  = (const float*)d_in[8];
    const float* bgz  = (const float*)d_in[9];
    const float* Wgh  = (const float*)d_in[10];
    const float* a1h  = (const float*)d_in[11];
    const float* a2h  = (const float*)d_in[12];
    const float* bgh  = (const float*)d_in[13];
    const float* gmm  = (const float*)d_in[14];
    const float* bta  = (const float*)d_in[15];
    const float* mu   = (const float*)d_in[16];
    const float* vr   = (const float*)d_in[17];

    const size_t nBTNC = (size_t)BB * TT * NN * CC;   // 6,291,456
    const size_t nBTHN = (size_t)BB * TT * HH * NN;   //   393,216

    char* w = (char*)d_ws;
    unsigned short* fzT = (unsigned short*)w; w += nBTNC * 2;
    unsigned short* fhT = (unsigned short*)w; w += nBTNC * 2;
    float* xwz = (float*)w; w += nBTNC * 4;
    float* xwh = (float*)w; w += nBTNC * 4;
    float* f1z = (float*)w; w += nBTHN * 4;
    float* f2z = (float*)w; w += nBTHN * 4;
    float* f1h = (float*)w; w += nBTHN * 4;
    float* f2h = (float*)w; w += nBTHN * 4;
    float* gz  = (float*)w; w += nBTNC * 4;
    float* gh  = (float*)w; w += nBTNC * 4;

    k_proj<<<dim3(NN / 128, BB * TT), 256, 0, stream>>>(
        X, Wz, Wh, Wgz, Wgh, fzT, fhT, xwz, xwh);
    k_f1f2<<<dim3((unsigned)(nBTHN / 256)), 256, 0, stream>>>(
        fzT, fhT, a1z, a2z, a1h, a2h, f1z, f2z, f1h, f2h);
    k_attn<<<dim3(NN / 128, BB * TT, 2 * HH), 256, 0, stream>>>(
        bias, fzT, fhT, f1z, f2z, f1h, f2h, bgz, bgh, gz, gh);
    k_recur<<<dim3((BB * NN * CC) / 256), 256, 0, stream>>>(
        gz, gh, xwz, xwh, Zb, Hb, gmm, bta, mu, vr, (float*)d_out);
}
